// Selfattention_90151363543520
// MI455X (gfx1250) — matrix operand order repaired and checked by reading, not compiled, not run on hardware
//
#include <hip/hip_runtime.h>
#include <hip/hip_bf16.h>

typedef __attribute__((ext_vector_type(16))) __bf16 v16bf;
typedef __attribute__((ext_vector_type(8)))  __bf16 v8bf;
typedef __attribute__((ext_vector_type(4)))  __bf16 v4bf;
typedef __attribute__((ext_vector_type(8)))  float  v8f;

union Frag16 { v16bf v; v8bf h[2]; };

#define BATCH  4
#define SEQ    2048
#define DMODEL 1024
#define BPROWS (BATCH * SEQ)       // 8192
#define TM 128
#define TN 128
#define TK 32
#define LSTR 40                    // bf16 per LDS row: 32 + 8 pad; 80B row stride (16B aligned)
#define SCALE 0.03125f             // 1/sqrt(1024)

// ---- CDNA5 async global->LDS copy (ASYNCcnt-tracked, no VGPR staging) -------
// ISA 10.2: generic LDS addresses map to LDS via addr[31:0], so the low 32 bits
// of a __shared__ pointer are the LDS byte offset the instruction needs.
__device__ __forceinline__ void async_copy16(const void* lds_dst, const void* gsrc) {
    unsigned           loff = (unsigned)(size_t)lds_dst;
    unsigned long long ga   = (unsigned long long)(size_t)gsrc;
    asm volatile("global_load_async_to_lds_b128 %0, %1, off"
                 :: "v"(loff), "v"(ga) : "memory");
}
__device__ __forceinline__ void wait_async0() {
    asm volatile("s_wait_asynccnt 0x0" ::: "memory");
}

// ---- WMMA fragment loads (wave32; ISA 7.12.2 layouts) -----------------------
// A (16x32 bf16): lane L holds row M=L&15; halves 0-7 = K[(L>=16?8:0)+h],
//                 halves 8-15 = K[16+(L>=16?8:0)+h].
__device__ __forceinline__ v16bf load_afrag(const __bf16* row, int lhalf) {
    Frag16 f;
    f.h[0] = *(const v8bf*)(row + lhalf * 8);
    f.h[1] = *(const v8bf*)(row + 16 + lhalf * 8);
    return f.v;
}
// B (32x16 bf16): lane L holds col N=L&15; halves 0-7 = K[(L>=16?8:0)+h],
//                 halves 8-15 = K[16+(L>=16?8:0)+h].
__device__ __forceinline__ v16bf load_bfrag(const __bf16* row, int lhalf) {
    Frag16 f;
    f.h[0] = *(const v8bf*)(row + lhalf * 8);
    f.h[1] = *(const v8bf*)(row + 16 + lhalf * 8);
    return f.v;
}

// ---- Kernel 1: Q/K/V projection  Y = bf16(x @ W^T + b) ----------------------
// Synchronous staging (needs fp32->bf16 conversion in transit).
__global__ __launch_bounds__(256)
void qkv_kernel(const float* __restrict__ x,
                const float* __restrict__ Wq, const float* __restrict__ bq,
                const float* __restrict__ Wk, const float* __restrict__ bk,
                const float* __restrict__ Wv, const float* __restrict__ bv,
                __bf16* __restrict__ Qg, __bf16* __restrict__ Kg, __bf16* __restrict__ Vg)
{
    __shared__ __align__(16) __bf16 As[TM][LSTR];
    __shared__ __align__(16) __bf16 Bs[TN][LSTR];

    const int which = blockIdx.z;
    const float* W    = (which == 0) ? Wq : (which == 1) ? Wk : Wv;
    const float* bias = (which == 0) ? bq : (which == 1) ? bk : bv;
    __bf16*      Y    = (which == 0) ? Qg : (which == 1) ? Kg : Vg;

    const int m0 = blockIdx.x * TM;
    const int n0 = blockIdx.y * TN;

    const int t = threadIdx.x;
    const int lane  = t & 31;
    const int wave  = t >> 5;
    const int wr    = wave & 3;
    const int wc    = wave >> 2;
    const int lhalf = lane >> 4;
    const int l15   = lane & 15;

    v8f acc[2][4];
    const v8f zero = {0.f,0.f,0.f,0.f,0.f,0.f,0.f,0.f};
#pragma unroll
    for (int mi = 0; mi < 2; ++mi)
#pragma unroll
        for (int ni = 0; ni < 4; ++ni) acc[mi][ni] = zero;

    for (int kb = 0; kb < DMODEL; kb += TK) {
#pragma unroll
        for (int j = 0; j < 4; ++j) {
            const int idx = t + 256 * j;     // 1024 float4 chunks
            const int row = idx >> 3;
            const int c4  = idx & 7;
            float4 fa = *(const float4*)(x + (size_t)(m0 + row) * DMODEL + kb + c4 * 4);
            __bf16* da = &As[row][c4 * 4];
            da[0] = (__bf16)fa.x; da[1] = (__bf16)fa.y;
            da[2] = (__bf16)fa.z; da[3] = (__bf16)fa.w;
            float4 fb = *(const float4*)(W + (size_t)(n0 + row) * DMODEL + kb + c4 * 4);
            __bf16* db = &Bs[row][c4 * 4];
            db[0] = (__bf16)fb.x; db[1] = (__bf16)fb.y;
            db[2] = (__bf16)fb.z; db[3] = (__bf16)fb.w;
        }
        __syncthreads();

        v16bf a[2], b[4];
#pragma unroll
        for (int mi = 0; mi < 2; ++mi)
            a[mi] = load_afrag(&As[wr * 32 + mi * 16 + l15][0], lhalf);
#pragma unroll
        for (int ni = 0; ni < 4; ++ni)
            b[ni] = load_bfrag(&Bs[wc * 64 + ni * 16 + l15][0], lhalf);
#pragma unroll
        for (int mi = 0; mi < 2; ++mi)
#pragma unroll
            for (int ni = 0; ni < 4; ++ni)
                acc[mi][ni] = __builtin_amdgcn_wmma_f32_16x16x32_bf16(
                    false, a[mi], false, b[ni], (short)0, acc[mi][ni], false, false);
        __syncthreads();
    }

#pragma unroll
    for (int ni = 0; ni < 4; ++ni) {
        const int n = n0 + wc * 64 + ni * 16 + l15;
        const float bb = bias[n];
#pragma unroll
        for (int mi = 0; mi < 2; ++mi) {
            const int mbase = m0 + wr * 32 + mi * 16 + lhalf * 8;
#pragma unroll
            for (int r = 0; r < 8; ++r)
                Y[(size_t)(mbase + r) * DMODEL + n] = (__bf16)(acc[mi][ni][r] + bb);
        }
    }
}

// ---- Kernel 2: S = (Q K^T) * scale, causal mask -> fp32 ---------------------
// Async double-buffered pipeline: one barrier per K-step, transfers overlap WMMA.
__global__ __launch_bounds__(256)
void scores_kernel(const __bf16* __restrict__ Qg, const __bf16* __restrict__ Kg,
                   float* __restrict__ Sg)
{
    const int k0 = blockIdx.x * TN;
    const int q0 = blockIdx.y * TM;
    const int bz = blockIdx.z;
    const int t  = threadIdx.x;

    float* Srow0 = Sg + ((size_t)bz * SEQ + q0) * SEQ;
    const float ninf = -__builtin_inff();

    if (k0 >= q0 + TM) {                     // tile fully above the diagonal
#pragma unroll
        for (int j = 0; j < 16; ++j) {
            const int idx = t + 256 * j;     // 4096 float4 chunks = 128x128
            const int row = idx >> 5;
            const int c4  = idx & 31;
            *(float4*)(Srow0 + (size_t)row * SEQ + k0 + c4 * 4) =
                make_float4(ninf, ninf, ninf, ninf);
        }
        return;
    }

    __shared__ __align__(16) __bf16 As[2][TM][LSTR];
    __shared__ __align__(16) __bf16 Bs[2][TN][LSTR];

    const int lane  = t & 31;
    const int wave  = t >> 5;
    const int wr    = wave & 3;
    const int wc    = wave >> 2;
    const int lhalf = lane >> 4;
    const int l15   = lane & 15;

    const __bf16* Qbase = Qg + ((size_t)bz * SEQ + q0) * DMODEL;
    const __bf16* Kbase = Kg + ((size_t)bz * SEQ + k0) * DMODEL;

    // per-thread staging coordinates: 2x A chunks + 2x B chunks of 16B per stage
    const int srow0 = (t + 0)   >> 2, sc0 = ((t + 0)   & 3) * 8;
    const int srow1 = (t + 256) >> 2, sc1 = ((t + 256) & 3) * 8;

#define SC_ISSUE(buf, kb)                                                           \
    do {                                                                            \
        async_copy16(&As[buf][srow0][sc0], Qbase + (size_t)srow0 * DMODEL + (kb) + sc0); \
        async_copy16(&Bs[buf][srow0][sc0], Kbase + (size_t)srow0 * DMODEL + (kb) + sc0); \
        async_copy16(&As[buf][srow1][sc1], Qbase + (size_t)srow1 * DMODEL + (kb) + sc1); \
        async_copy16(&Bs[buf][srow1][sc1], Kbase + (size_t)srow1 * DMODEL + (kb) + sc1); \
    } while (0)

    v8f acc[2][4];
    const v8f zero = {0.f,0.f,0.f,0.f,0.f,0.f,0.f,0.f};
#pragma unroll
    for (int mi = 0; mi < 2; ++mi)
#pragma unroll
        for (int ni = 0; ni < 4; ++ni) acc[mi][ni] = zero;

    SC_ISSUE(0, 0);
    int buf = 0;
    for (int kb = 0; kb < DMODEL; kb += TK) {
        wait_async0();                       // stage kb landed (this wave)
        __syncthreads();                     // all waves: data ready, prev reads done
        if (kb + TK < DMODEL) SC_ISSUE(buf ^ 1, kb + TK);   // overlaps compute below

        v16bf a[2], b[4];
#pragma unroll
        for (int mi = 0; mi < 2; ++mi)
            a[mi] = load_afrag(&As[buf][wr * 32 + mi * 16 + l15][0], lhalf);
#pragma unroll
        for (int ni = 0; ni < 4; ++ni)
            b[ni] = load_bfrag(&Bs[buf][wc * 64 + ni * 16 + l15][0], lhalf);
#pragma unroll
        for (int mi = 0; mi < 2; ++mi)
#pragma unroll
            for (int ni = 0; ni < 4; ++ni)
                acc[mi][ni] = __builtin_amdgcn_wmma_f32_16x16x32_bf16(
                    false, a[mi], false, b[ni], (short)0, acc[mi][ni], false, false);
        buf ^= 1;
    }
#undef SC_ISSUE

#pragma unroll
    for (int ni = 0; ni < 4; ++ni) {
        const int kk = k0 + wc * 64 + ni * 16 + l15;
#pragma unroll
        for (int mi = 0; mi < 2; ++mi) {
            const int qloc = wr * 32 + mi * 16 + lhalf * 8;
#pragma unroll
            for (int r = 0; r < 8; ++r) {
                const int q = q0 + qloc + r;
                const float v = acc[mi][ni][r] * SCALE;
                Srow0[(size_t)(qloc + r) * SEQ + kk] = (kk > q) ? ninf : v;
            }
        }
    }
}

// ---- Kernel 3: row softmax, fp32 -> bf16 in place ---------------------------
__global__ __launch_bounds__(256)
void softmax_kernel(float* __restrict__ Sg)
{
    const int t    = threadIdx.x;
    const int lane = t & 31;
    const int wave = t >> 5;
    float* rp = Sg + (size_t)blockIdx.x * SEQ;

    float v[8];
    float mx = -__builtin_inff();
#pragma unroll
    for (int j = 0; j < 8; ++j) { v[j] = rp[t + 256 * j]; mx = fmaxf(mx, v[j]); }
#pragma unroll
    for (int o = 16; o > 0; o >>= 1) mx = fmaxf(mx, __shfl_xor(mx, o, 32));

    __shared__ float red[8];
    if (lane == 0) red[wave] = mx;
    __syncthreads();
    mx = red[0];
#pragma unroll
    for (int w = 1; w < 8; ++w) mx = fmaxf(mx, red[w]);

    float s = 0.f;
#pragma unroll
    for (int j = 0; j < 8; ++j) { v[j] = __expf(v[j] - mx); s += v[j]; }
#pragma unroll
    for (int o = 16; o > 0; o >>= 1) s += __shfl_xor(s, o, 32);
    __syncthreads();
    if (lane == 0) red[wave] = s;
    __syncthreads();
    s = 0.f;
#pragma unroll
    for (int w = 0; w < 8; ++w) s += red[w];
    const float inv = 1.0f / s;

    // all reads of this row completed before the first barrier -> in-place OK
    __bf16* op = (__bf16*)rp;
#pragma unroll
    for (int j = 0; j < 8; ++j) op[t + 256 * j] = (__bf16)(v[j] * inv);
}

// ---- Kernel 4: out = P @ V (fp32 out), causal k bound -----------------------
// P tile: async double-buffered.  V tile: synchronous transpose staging,
// double-buffered in the same pipeline slot (next __syncthreads' dscnt-wait
// guarantees cross-wave visibility).
__global__ __launch_bounds__(256)
void pv_kernel(const float* __restrict__ Sg, const __bf16* __restrict__ Vg,
               float* __restrict__ out)
{
    const int mrow0 = blockIdx.x * TM;       // tiles never cross batch boundary
    const int b  = mrow0 / SEQ;
    const int q0 = mrow0 % SEQ;
    const int d0 = blockIdx.y * TN;

    __shared__ __align__(16) __bf16 As[2][TM][LSTR];
    __shared__ __align__(16) __bf16 Vs[2][TN][LSTR];   // Vs[d][k]

    const int t = threadIdx.x;
    const int lane  = t & 31;
    const int wave  = t >> 5;
    const int wr    = wave & 3;
    const int wc    = wave >> 2;
    const int lhalf = lane >> 4;
    const int l15   = lane & 15;

    const __bf16* Pbase = (const __bf16*)(Sg + ((size_t)b * SEQ + q0) * SEQ);
    const size_t  prstride = (size_t)SEQ * 2;  // bf16 elements between P rows (rows are fp32-sized)
    const __bf16* Vbase = Vg + (size_t)b * SEQ * DMODEL + d0;

    const int srow0 = (t + 0)   >> 2, sc0 = ((t + 0)   & 3) * 8;
    const int srow1 = (t + 256) >> 2, sc1 = ((t + 256) & 3) * 8;

#define PV_ISSUE_A(buf, kb)                                                          \
    do {                                                                             \
        async_copy16(&As[buf][srow0][sc0], Pbase + (size_t)srow0 * prstride + (kb) + sc0); \
        async_copy16(&As[buf][srow1][sc1], Pbase + (size_t)srow1 * prstride + (kb) + sc1); \
    } while (0)

#define PV_STAGE_V(buf, kb)                                                          \
    _Pragma("unroll")                                                                \
    for (int j = 0; j < 4; ++j) {                                                    \
        const int idx = t + 256 * j;   /* 1024 v4bf chunks */                        \
        const int k   = idx >> 5;                                                    \
        const int d4  = idx & 31;                                                    \
        v4bf vv = *(const v4bf*)(Vbase + (size_t)((kb) + k) * DMODEL + d4 * 4);      \
        Vs[buf][d4 * 4 + 0][k] = vv[0];                                              \
        Vs[buf][d4 * 4 + 1][k] = vv[1];                                              \
        Vs[buf][d4 * 4 + 2][k] = vv[2];                                              \
        Vs[buf][d4 * 4 + 3][k] = vv[3];                                              \
    }

    v8f acc[2][4];
    const v8f zero = {0.f,0.f,0.f,0.f,0.f,0.f,0.f,0.f};
#pragma unroll
    for (int mi = 0; mi < 2; ++mi)
#pragma unroll
        for (int ni = 0; ni < 4; ++ni) acc[mi][ni] = zero;

    const int kend = q0 + TM;                // causal: this tile only needs k <= q0+127
    PV_ISSUE_A(0, 0);
    PV_STAGE_V(0, 0);
    int buf = 0;
    for (int kb = 0; kb < kend; kb += TK) {
        wait_async0();
        __syncthreads();
        if (kb + TK < kend) {
            PV_ISSUE_A(buf ^ 1, kb + TK);
            PV_STAGE_V(buf ^ 1, kb + TK);
        }

        v16bf a[2], bfrag[4];
#pragma unroll
        for (int mi = 0; mi < 2; ++mi)
            a[mi] = load_afrag(&As[buf][wr * 32 + mi * 16 + l15][0], lhalf);
#pragma unroll
        for (int ni = 0; ni < 4; ++ni)
            bfrag[ni] = load_bfrag(&Vs[buf][wc * 64 + ni * 16 + l15][0], lhalf);
#pragma unroll
        for (int mi = 0; mi < 2; ++mi)
#pragma unroll
            for (int ni = 0; ni < 4; ++ni)
                acc[mi][ni] = __builtin_amdgcn_wmma_f32_16x16x32_bf16(
                    false, a[mi], false, bfrag[ni], (short)0, acc[mi][ni], false, false);
        buf ^= 1;
    }
#undef PV_ISSUE_A
#undef PV_STAGE_V

#pragma unroll
    for (int ni = 0; ni < 4; ++ni) {
        const int d = d0 + wc * 64 + ni * 16 + l15;
#pragma unroll
        for (int mi = 0; mi < 2; ++mi) {
            const int q = q0 + wr * 32 + mi * 16 + lhalf * 8;
#pragma unroll
            for (int r = 0; r < 8; ++r)
                out[((size_t)b * SEQ + q + r) * DMODEL + d] = acc[mi][ni][r];
        }
    }
}

// ---- host launch ------------------------------------------------------------
extern "C" void kernel_launch(void* const* d_in, const int* in_sizes, int n_in,
                              void* d_out, int out_size, void* d_ws, size_t ws_size,
                              hipStream_t stream)
{
    const float* x  = (const float*)d_in[0];
    const float* Wq = (const float*)d_in[1];
    const float* bq = (const float*)d_in[2];
    const float* Wk = (const float*)d_in[3];
    const float* bk = (const float*)d_in[4];
    const float* Wv = (const float*)d_in[5];
    const float* bv = (const float*)d_in[6];
    float* out = (float*)d_out;

    char* ws = (char*)d_ws;
    __bf16* Qg = (__bf16*)(ws);                          // 16 MiB
    __bf16* Kg = (__bf16*)(ws + ((size_t)16 << 20));     // 16 MiB
    __bf16* Vg = (__bf16*)(ws + ((size_t)32 << 20));     // 16 MiB
    float*  Sg = (float*)(ws + ((size_t)48 << 20));      // 64 MiB (scores, then bf16 P in place)

    const dim3 blk(256);
    qkv_kernel<<<dim3(BPROWS / TM, DMODEL / TN, 3), blk, 0, stream>>>(
        x, Wq, bq, Wk, bk, Wv, bv, Qg, Kg, Vg);
    scores_kernel<<<dim3(SEQ / TN, SEQ / TM, BATCH), blk, 0, stream>>>(Qg, Kg, Sg);
    softmax_kernel<<<dim3(BPROWS), blk, 0, stream>>>(Sg);
    pv_kernel<<<dim3(BPROWS / TM, DMODEL / TN), blk, 0, stream>>>(Sg, Vg, out);
}
